// copied_Llama4TextDecoderLayer_70102456205609
// MI455X (gfx1250) — matrix operand order repaired and checked by reading, not compiled, not run on hardware
//
#include <hip/hip_runtime.h>
#include <math.h>

// ---------------------------------------------------------------------------
// CDNA5 (gfx1250) wave32 WMMA types
// ---------------------------------------------------------------------------
typedef __attribute__((ext_vector_type(16))) __bf16 v16bf;
typedef __attribute__((ext_vector_type(8)))  float  v8f;

union Frag {             // 16 bf16 striped per-lane = 8 VGPRs
    v16bf     v;
    unsigned  u[8];
};

__device__ inline float half16_max(float x) {
#pragma unroll
    for (int m = 1; m < 16; m <<= 1) x = fmaxf(x, __shfl_xor(x, m, 32));
    return x;
}
__device__ inline float half16_sum(float x) {
#pragma unroll
    for (int m = 1; m < 16; m <<= 1) x += __shfl_xor(x, m, 32);
    return x;
}

// ---------------------------------------------------------------------------
// RMSNorm: one block per row of D=4096
// ---------------------------------------------------------------------------
__global__ void rmsnorm_kernel(const float* __restrict__ x,
                               const float* __restrict__ w,
                               float* __restrict__ out, int D) {
    __shared__ float red[256];
    const int row = blockIdx.x;
    const float* xr = x + (size_t)row * D;
    float s = 0.f;
    for (int j = threadIdx.x; j < D; j += 256) { float v = xr[j]; s += v * v; }
    red[threadIdx.x] = s;
    __syncthreads();
    for (int off = 128; off > 0; off >>= 1) {
        if (threadIdx.x < off) red[threadIdx.x] += red[threadIdx.x + off];
        __syncthreads();
    }
    const float r = rsqrtf(red[0] / (float)D + 1e-5f);
    float* orow = out + (size_t)row * D;
    for (int j = threadIdx.x; j < D; j += 256) orow[j] = xr[j] * r * w[j];
}

// ---------------------------------------------------------------------------
// Tiled bf16-WMMA GEMM:  C[M,N] = A[M,K] * B[K,N]  (A,B fp32 row-major)
//   block tile 128x128, K-step 32, 256 threads = 8 waves (4 in M x 2 in N)
//   each wave: 32x64 region = 2x4 tiles of v_wmma_f32_16x16x32_bf16
//   Double-buffered LDS + software-pipelined global loads: issue K-step kt+1
//   global_load_b128s before computing kt from LDS; one barrier per K-step.
//   grid: blockIdx.x = M-block (fastest) so the 16 M-blocks sharing one
//   128-col weight stripe (2 MB, L2-resident) run back-to-back -> each
//   weight byte streams from HBM exactly once.
// MODE 0: C = acc      MODE 1: C = acc + aux      MODE 2: C = silu(aux)*acc
// ---------------------------------------------------------------------------
template <int MODE>
__global__ __launch_bounds__(256, 1) void gemm_bf16(
        const float* __restrict__ A,
        const float* __restrict__ B,
        float* __restrict__ C,
        int M, int N, int K,
        const float* __restrict__ aux) {
    constexpr int BM = 128, BN = 128, BK = 32;
    __shared__ __bf16 As[2][BM][BK];   // row-major, k pairs contiguous
    __shared__ __bf16 Bs[2][BN][BK];   // transposed: [n][k]

    const int tid  = threadIdx.x;
    const int wave = tid >> 5;
    const int lane = tid & 31;
    const int lr   = lane & 15;
    const int lh   = lane >> 4;
    const int wm   = wave >> 1;                // 0..3 -> 32-row strips
    const int wn   = wave & 1;                 // 0..1 -> 64-col strips

    const int rowBase = blockIdx.x * BM;       // M fastest
    const int colBase = blockIdx.y * BN;

    // per-thread staging coordinates (4 float4 pieces each for A and B)
    const int aRow = tid >> 3;                 // 0..31 (+32 per piece)
    const int aCol = (tid & 7) << 2;           // 0..28
    const int bRow = tid >> 5;                 // 0..7  (+8 per piece)
    const int bCol = (tid & 31) << 2;          // 0..124

    float4 ra[4], rb[4];
    auto loadTiles = [&](int kt) {
#pragma unroll
        for (int j = 0; j < 4; ++j) {
            ra[j] = *(const float4*)(A + (size_t)(rowBase + aRow + 32 * j) * K + kt * BK + aCol);
            rb[j] = *(const float4*)(B + (size_t)(kt * BK + bRow + 8 * j) * N + colBase + bCol);
        }
    };
    auto storeTiles = [&](int buf) {
#pragma unroll
        for (int j = 0; j < 4; ++j) {
            As[buf][aRow + 32 * j][aCol + 0] = (__bf16)ra[j].x;
            As[buf][aRow + 32 * j][aCol + 1] = (__bf16)ra[j].y;
            As[buf][aRow + 32 * j][aCol + 2] = (__bf16)ra[j].z;
            As[buf][aRow + 32 * j][aCol + 3] = (__bf16)ra[j].w;
            Bs[buf][bCol + 0][bRow + 8 * j]  = (__bf16)rb[j].x;
            Bs[buf][bCol + 1][bRow + 8 * j]  = (__bf16)rb[j].y;
            Bs[buf][bCol + 2][bRow + 8 * j]  = (__bf16)rb[j].z;
            Bs[buf][bCol + 3][bRow + 8 * j]  = (__bf16)rb[j].w;
        }
    };

    v8f acc[2][4] = {};

    const int ksteps = K / BK;
    loadTiles(0);
    storeTiles(0);
    __syncthreads();

    int p = 0;
    for (int kt = 0; kt < ksteps; ++kt) {
        const bool more = (kt + 1) < ksteps;
        if (more) loadTiles(kt + 1);           // overlap HBM with WMMA below

        // --- A fragments (16x32, contiguous k = 16*lh + s, same order as B)
        Frag a[2];
#pragma unroll
        for (int mt = 0; mt < 2; ++mt) {
            const int row = wm * 32 + mt * 16 + lr;
#pragma unroll
            for (int v = 0; v < 8; ++v) {
                const int k0 = (lh << 4) + (v << 1);
                a[mt].u[v] = *(const unsigned*)&As[p][row][k0];
            }
        }
        // --- B fragments + WMMA
#pragma unroll
        for (int nt = 0; nt < 4; ++nt) {
            Frag b;
            const int col = wn * 64 + nt * 16 + lr;
#pragma unroll
            for (int v = 0; v < 8; ++v)
                b.u[v] = *(const unsigned*)&Bs[p][col][(lh << 4) + (v << 1)];
#pragma unroll
            for (int mt = 0; mt < 2; ++mt)
                acc[mt][nt] = __builtin_amdgcn_wmma_f32_16x16x32_bf16(
                    false, a[mt].v, false, b.v, (short)0, acc[mt][nt], false, false);
        }

        if (more) {
            storeTiles(p ^ 1);
            __syncthreads();
            p ^= 1;
        }
    }

    // --- epilogue (C/D layout: VGPR r -> M = r + 8*lh, lane -> N)
#pragma unroll
    for (int mt = 0; mt < 2; ++mt)
#pragma unroll
        for (int nt = 0; nt < 4; ++nt)
#pragma unroll
            for (int r = 0; r < 8; ++r) {
                const int grow = rowBase + wm * 32 + mt * 16 + r + (lh << 3);
                const int gcol = colBase + wn * 64 + nt * 16 + lr;
                const size_t idx = (size_t)grow * N + gcol;
                float val = acc[mt][nt][r];
                if (MODE == 1) {
                    val += aux[idx];
                } else if (MODE == 2) {
                    const float g = aux[idx];
                    val *= g / (1.f + __expf(-g));   // silu(gate) * up
                }
                C[idx] = val;
            }
}

// ---------------------------------------------------------------------------
// Fused RoPE + L2-norm over each 128-wide head of q (32 heads) and k (8 heads)
// one block (64 threads) per (seq position, head)
// ---------------------------------------------------------------------------
__global__ void rope_l2_kernel(float* __restrict__ q, float* __restrict__ k,
                               const float* __restrict__ cosb,
                               const float* __restrict__ sinb) {
    __shared__ float red[64];
    const int s  = blockIdx.x / 40;
    const int hh = blockIdx.x % 40;
    float* base = (hh < 32) ? (q + (size_t)s * 4096 + hh * 128)
                            : (k + (size_t)s * 1024 + (hh - 32) * 128);
    const int i = threadIdx.x;                 // pair index 0..63
    const float xr = base[2 * i], xi = base[2 * i + 1];
    const float c  = cosb[(size_t)s * 64 + i];
    const float sn = sinb[(size_t)s * 64 + i];
    const float orr = xr * c - xi * sn;
    const float oii = xr * sn + xi * c;
    red[i] = orr * orr + oii * oii;
    __syncthreads();
    for (int off = 32; off > 0; off >>= 1) {
        if (i < off) red[i] += red[i + off];
        __syncthreads();
    }
    const float r = rsqrtf(red[0] / 128.f + 1e-5f);
    base[2 * i]     = orr * r;
    base[2 * i + 1] = oii * r;
}

// ---------------------------------------------------------------------------
// Flash attention (causal, GQA 4:1), all-WMMA.
// grid = (S/16 q-tiles, 8 kv heads), block = 128 thr = 4 waves (4 q-heads)
// Q tile lives in LDS so persistent VGPR state is just the 64-VGPR output
// accumulator + softmax stats -> no scratch spills.
// ---------------------------------------------------------------------------
__global__ __launch_bounds__(128, 1) void flash_attn_kernel(
        const float* __restrict__ Q,
        const float* __restrict__ K,
        const float* __restrict__ V,
        float* __restrict__ O) {
    __shared__ __bf16 Qs[4][16][128];  // per-wave q tile
    __shared__ __bf16 Ks[32][128];     // [key][d]
    __shared__ __bf16 Vt[128][32];     // [d][key] (transposed)
    __shared__ __bf16 Ps[4][16][32];   // per-wave P tile

    const int tid   = threadIdx.x;
    const int wave  = tid >> 5;
    const int lane  = tid & 31;
    const int lr    = lane & 15;
    const int lh    = lane >> 4;
    const int qtile = blockIdx.x;
    const int kvh   = blockIdx.y;
    const int head  = kvh * 4 + wave;
    const int q0    = qtile * 16;
    const float scaling = 0.08838834764831845f;   // 128^-0.5

    // --- stage this wave's 16x128 q tile into LDS (bf16)
#pragma unroll
    for (int i = lane; i < (16 * 128) / 4; i += 32) {
        const int row = i >> 5;
        const int col = (i & 31) << 2;
        const float4 f = *(const float4*)(Q + (size_t)(q0 + row) * 4096 + head * 128 + col);
        Qs[wave][row][col + 0] = (__bf16)f.x;
        Qs[wave][row][col + 1] = (__bf16)f.y;
        Qs[wave][row][col + 2] = (__bf16)f.z;
        Qs[wave][row][col + 3] = (__bf16)f.w;
    }

    v8f   o[8] = {};
    float m[8], l[8];
#pragma unroll
    for (int r = 0; r < 8; ++r) { m[r] = -3.0e38f; l[r] = 0.f; }

    const int nch = (q0 + 15) / 32 + 1;        // causal: stop at diagonal
    for (int ch = 0; ch < nch; ++ch) {
        __syncthreads();
        // stage K chunk (32x128) and V chunk (transposed)
#pragma unroll
        for (int i = tid; i < (32 * 128) / 4; i += 128) {
            const int row = i >> 5;
            const int col = (i & 31) << 2;
            const float4 f = *(const float4*)(K + (size_t)(ch * 32 + row) * 1024 + kvh * 128 + col);
            Ks[row][col + 0] = (__bf16)f.x;
            Ks[row][col + 1] = (__bf16)f.y;
            Ks[row][col + 2] = (__bf16)f.z;
            Ks[row][col + 3] = (__bf16)f.w;
            const float4 g = *(const float4*)(V + (size_t)(ch * 32 + row) * 1024 + kvh * 128 + col);
            Vt[col + 0][row] = (__bf16)g.x;
            Vt[col + 1][row] = (__bf16)g.y;
            Vt[col + 2][row] = (__bf16)g.z;
            Vt[col + 3][row] = (__bf16)g.w;
        }
        __syncthreads();

        // ---- scores: two 16x16 tiles (32 keys), accumulate over 4 d-chunks
        v8f s0 = {}, s1 = {};
#pragma unroll
        for (int c = 0; c < 4; ++c) {
            Frag aq, b0, b1;
#pragma unroll
            for (int v = 0; v < 8; ++v) {
                const int k0 = (lh << 4) + (v << 1);
                aq.u[v] = *(const unsigned*)&Qs[wave][lr][c * 32 + k0];
                const int kk = c * 32 + (lh << 4) + (v << 1);
                b0.u[v] = *(const unsigned*)&Ks[lr][kk];
                b1.u[v] = *(const unsigned*)&Ks[16 + lr][kk];
            }
            s0 = __builtin_amdgcn_wmma_f32_16x16x32_bf16(false, aq.v, false, b0.v, (short)0, s0, false, false);
            s1 = __builtin_amdgcn_wmma_f32_16x16x32_bf16(false, aq.v, false, b1.v, (short)0, s1, false, false);
        }

        // ---- online softmax (row-wise across the 16-lane half)
#pragma unroll
        for (int r = 0; r < 8; ++r) {
            const int grow = q0 + r + (lh << 3);
            const int k0g  = ch * 32 + lr;
            const float x0 = s0[r] * scaling + ((k0g      > grow) ? -1e9f : 0.f);
            const float x1 = s1[r] * scaling + ((k0g + 16 > grow) ? -1e9f : 0.f);
            const float mx = half16_max(fmaxf(x0, x1));
            const float nm = fmaxf(m[r], mx);
            const float sc = __expf(m[r] - nm);
            const float p0 = __expf(x0 - nm);
            const float p1 = __expf(x1 - nm);
            l[r] = l[r] * sc + half16_sum(p0 + p1);
            m[r] = nm;
#pragma unroll
            for (int nt = 0; nt < 8; ++nt) o[nt][r] *= sc;
            Ps[wave][r + (lh << 3)][lr]      = (__bf16)p0;
            Ps[wave][r + (lh << 3)][16 + lr] = (__bf16)p1;
        }

        // ---- P (16x32) x V (32x128) -> o, 8 N-tiles
        Frag pa;
#pragma unroll
        for (int v = 0; v < 8; ++v) {
            const int k0 = (lh << 4) + (v << 1);
            pa.u[v] = *(const unsigned*)&Ps[wave][lr][k0];
        }
#pragma unroll
        for (int nt = 0; nt < 8; ++nt) {
            Frag vb;
#pragma unroll
            for (int v = 0; v < 8; ++v)
                vb.u[v] = *(const unsigned*)&Vt[nt * 16 + lr][(lh << 4) + (v << 1)];
            o[nt] = __builtin_amdgcn_wmma_f32_16x16x32_bf16(false, pa.v, false, vb.v, (short)0, o[nt], false, false);
        }
    }

    // ---- normalize and write [S, H*128]
#pragma unroll
    for (int r = 0; r < 8; ++r) {
        const float inv = 1.f / l[r];
        const size_t rowoff = (size_t)(q0 + r + (lh << 3)) * 4096 + head * 128;
#pragma unroll
        for (int nt = 0; nt < 8; ++nt)
            O[rowoff + nt * 16 + lr] = o[nt][r] * inv;
    }
}

// ---------------------------------------------------------------------------
// Host-side launch sequence
// ---------------------------------------------------------------------------
extern "C" void kernel_launch(void* const* d_in, const int* in_sizes, int n_in,
                              void* d_out, int out_size, void* d_ws, size_t ws_size,
                              hipStream_t stream) {
    (void)in_sizes; (void)n_in; (void)out_size; (void)ws_size;
    const float* x     = (const float*)d_in[0];
    const float* fcos  = (const float*)d_in[1];
    const float* fsin  = (const float*)d_in[2];
    // d_in[3]: attention_mask (pure causal -> computed analytically)
    const float* ln1   = (const float*)d_in[4];
    const float* ln2   = (const float*)d_in[5];
    const float* wq    = (const float*)d_in[6];
    const float* wk    = (const float*)d_in[7];
    const float* wv    = (const float*)d_in[8];
    const float* wo    = (const float*)d_in[9];
    const float* wg    = (const float*)d_in[10];
    const float* wu    = (const float*)d_in[11];
    const float* wd    = (const float*)d_in[12];
    float* out = (float*)d_out;

    const int S = 2048, D = 4096, I = 16384, KVD = 1024;

    float* ws   = (float*)d_ws;
    size_t off  = 0;
    float* hs   = ws + off; off += (size_t)S * D;    // normed (reused for both norms)
    float* q    = ws + off; off += (size_t)S * D;
    float* kbuf = ws + off; off += (size_t)S * KVD;
    float* vbuf = ws + off; off += (size_t)S * KVD;
    float* attn = ws + off; off += (size_t)S * D;
    float* res2 = ws + off; off += (size_t)S * D;
    float* gate = ws + off; off += (size_t)S * I;    // then act in-place

    // 1. pre-attention RMSNorm
    rmsnorm_kernel<<<S, 256, 0, stream>>>(x, ln1, hs, D);
    // 2. q/k/v projections (grid: M fastest for weight L2 reuse)
    gemm_bf16<0><<<dim3(S / 128, D / 128),   256, 0, stream>>>(hs, wq, q,    S, D,   D, nullptr);
    gemm_bf16<0><<<dim3(S / 128, KVD / 128), 256, 0, stream>>>(hs, wk, kbuf, S, KVD, D, nullptr);
    gemm_bf16<0><<<dim3(S / 128, KVD / 128), 256, 0, stream>>>(hs, wv, vbuf, S, KVD, D, nullptr);
    // 3. RoPE + L2-norm on q and k
    rope_l2_kernel<<<S * 40, 64, 0, stream>>>(q, kbuf, fcos, fsin);
    // 4. causal flash attention
    flash_attn_kernel<<<dim3(S / 16, 8), 128, 0, stream>>>(q, kbuf, vbuf, attn);
    // 5. output projection + residual
    gemm_bf16<1><<<dim3(S / 128, D / 128), 256, 0, stream>>>(attn, wo, res2, S, D, D, x);
    // 6. post-attention RMSNorm
    rmsnorm_kernel<<<S, 256, 0, stream>>>(res2, ln2, hs, D);
    // 7. MLP: gate, then up fused with silu(gate)*up (act written over gate)
    gemm_bf16<0><<<dim3(S / 128, I / 128), 256, 0, stream>>>(hs, wg, gate, S, I, D, nullptr);
    gemm_bf16<2><<<dim3(S / 128, I / 128), 256, 0, stream>>>(hs, wu, gate, S, I, D, gate);
    // 8. down projection + residual -> output
    gemm_bf16<1><<<dim3(S / 128, D / 128), 256, 0, stream>>>(gate, wd, out, S, D, I, res2);
}
